// GuidedUpsampleConv1x1_72782515798654
// MI455X (gfx1250) — matrix operand order repaired and checked by reading, not compiled, not run on hardware
//
#include <hip/hip_runtime.h>
#include <math.h>

// ---------------------------------------------------------------------------
// GuidedUpsampleConv1x1 for MI455X (gfx1250, wave32, WMMA + async LDS DMA)
//
// Shapes:
//   x     [4, 512, 64, 128]   (B, Ci, h, w)   h*w = 8192
//   guide [4, 64, 128, 256]   (B, Cg, H, W)   H*W = 32768
//   w     [256, 512]          (Co, Ci)
//   gamma/beta [256]
//   out   [4, 256, 128, 256]
//
// Workspace (floats):
//   xv    [4][256][8192]  @ 0            (8,388,608)
//   scale [256]           @ 8,388,608
//   shift [256]           @ 8,388,864
//   attn  [4][9][32768]   @ 8,389,120    (1,179,648)
// ---------------------------------------------------------------------------

typedef __attribute__((ext_vector_type(16))) __bf16 v16bf;
typedef __attribute__((ext_vector_type(8)))  float  v8f;

__device__ __forceinline__ void bf16_split(float f, __bf16& hi, __bf16& lo) {
    // hi = truncate-to-bf16(f) (exactly representable), lo = RNE(f - hi).
    float hf = __uint_as_float(__float_as_uint(f) & 0xffff0000u);
    hi = (__bf16)hf;
    lo = (__bf16)(f - hf);
}

// Build A fragment (16-bit A 16x32 layout) from one w row:
// elements 0..7 -> K = base..base+7, elements 8..15 -> K = base+16..base+23.
__device__ __forceinline__ void load_a_frag(const float* __restrict__ wr,
                                            v16bf& hi, v16bf& lo) {
    float4 q0 = *(const float4*)(wr + 0);
    float4 q1 = *(const float4*)(wr + 4);
    float4 q2 = *(const float4*)(wr + 16);
    float4 q3 = *(const float4*)(wr + 20);
    float af[16] = { q0.x, q0.y, q0.z, q0.w, q1.x, q1.y, q1.z, q1.w,
                     q2.x, q2.y, q2.z, q2.w, q3.x, q3.y, q3.z, q3.w };
#pragma unroll
    for (int e = 0; e < 16; ++e) {
        __bf16 h, l; bf16_split(af[e], h, l);
        hi[e] = h; lo[e] = l;
    }
}

// gfx1250 async DMA: global -> LDS, tracked by ASYNCcnt (no VGPR data path).
// LDS address operand = low 32 bits of the generic shared pointer.
__device__ __forceinline__ void async_load_f32_to_lds(const float* gsrc,
                                                      const float* ldst) {
    unsigned lds = (unsigned)(uintptr_t)ldst;
    asm volatile("global_load_async_to_lds_b32 %0, %1, off th:TH_LOAD_NT"
                 :: "v"(lds), "v"(gsrc) : "memory");
}
__device__ __forceinline__ void wait_async_lds() {
    asm volatile("s_wait_asynccnt 0x0" ::: "memory");
}

// ---------------------------------------------------------------------------
// K1: xv[b][m][p] = sum_k w[m][k] * x[b][k][p]
// Block = 8 waves, one 16-pixel n-tile, all 256 output channels.
// Each wave owns two 16x16 m-tiles and shares one B fragment across both.
// x tile (32K x 16N f32) staged in LDS via async DMA, double-buffered.
// Error-compensated bf16 split -> 6 x v_wmma_f32_16x16x32_bf16 per K-step.
// ---------------------------------------------------------------------------
__global__ __launch_bounds__(256) void k1_gemm_wmma(
    const float* __restrict__ x, const float* __restrict__ w,
    float* __restrict__ xv)
{
    // [buf][pixel-col][K-row + pad]; stride 36 -> conflict-free fills/reads
    __shared__ __align__(16) float xs[2][16][36];

    const int tid  = threadIdx.x;
    const int lane = tid & 31;
    const int wave = tid >> 5;

    const int n0 = blockIdx.x << 4;                    // global pixel-tile base
    const int b  = n0 >> 13;                           // 8192 pixels per batch
    const int p0 = n0 & 8191;
    const float* __restrict__ xb = x + (size_t)b * 512u * 8192u + p0;

    const int lrow = tid >> 4;                         // 0..15 (K row pair)
    const int lcol = tid & 15;                         // pixel column

    // async preload of K rows 0..31 into buffer 0 (x is read exactly once)
    async_load_f32_to_lds(xb + (size_t)lrow * 8192u + lcol,        &xs[0][lcol][lrow]);
    async_load_f32_to_lds(xb + (size_t)(lrow + 16) * 8192u + lcol, &xs[0][lcol][lrow + 16]);

    const int m0      = wave << 5;                     // two m-tiles: m0, m0+16
    const int mrow    = m0 + (lane & 15);
    const int khalf_a = (lane >> 4) * 8;               // A: lanes 16-31 -> K+8
    const int khalf_b = (lane >> 4) * 8;               // B: lanes 16-31 -> K+8

    v8f acc0 = {}, acc1 = {};
    int buf = 0;
    for (int k0 = 0; k0 < 512; k0 += 32, buf ^= 1) {
        wait_async_lds();      // my fills of xs[buf] have landed in LDS
        __syncthreads();       // everyone's fills visible workgroup-wide
        if (k0 + 32 < 512) {   // kick off DMA for the next K tile
            async_load_f32_to_lds(xb + (size_t)(k0 + 32 + lrow) * 8192u + lcol,
                                  &xs[buf ^ 1][lcol][lrow]);
            async_load_f32_to_lds(xb + (size_t)(k0 + 48 + lrow) * 8192u + lcol,
                                  &xs[buf ^ 1][lcol][lrow + 16]);
        }
        if (k0 + 64 < 512)     // keep L2 warm two tiles ahead
            __builtin_prefetch(xb + (size_t)(k0 + 64 + lrow) * 8192u + lcol, 0, 3);

        // ---- B fragment from LDS (K = base..base+7, base+16..base+23 per lane-half) ----
        const float* __restrict__ bsrc = &xs[buf][lane & 15][khalf_b];
        float bf[16];
        *(float4*)(bf + 0)  = *(const float4*)(bsrc + 0);
        *(float4*)(bf + 4)  = *(const float4*)(bsrc + 4);
        *(float4*)(bf + 8)  = *(const float4*)(bsrc + 16);
        *(float4*)(bf + 12) = *(const float4*)(bsrc + 20);
        v16bf bh, bl;
#pragma unroll
        for (int e = 0; e < 16; ++e) {
            __bf16 h, l; bf16_split(bf[e], h, l);
            bh[e] = h; bl[e] = l;
        }

        // ---- A fragments for both m-tiles ----
        v16bf ah0, al0, ah1, al1;
        load_a_frag(w + (size_t)mrow * 512u + k0 + khalf_a,        ah0, al0);
        load_a_frag(w + (size_t)(mrow + 16) * 512u + k0 + khalf_a, ah1, al1);

        // ---- 3-term compensated products, B fragment reused twice ----
        acc0 = __builtin_amdgcn_wmma_f32_16x16x32_bf16(false, ah0, false, bh,
                                                       (short)0, acc0, false, false);
        acc0 = __builtin_amdgcn_wmma_f32_16x16x32_bf16(false, ah0, false, bl,
                                                       (short)0, acc0, false, false);
        acc0 = __builtin_amdgcn_wmma_f32_16x16x32_bf16(false, al0, false, bh,
                                                       (short)0, acc0, false, false);
        acc1 = __builtin_amdgcn_wmma_f32_16x16x32_bf16(false, ah1, false, bh,
                                                       (short)0, acc1, false, false);
        acc1 = __builtin_amdgcn_wmma_f32_16x16x32_bf16(false, ah1, false, bl,
                                                       (short)0, acc1, false, false);
        acc1 = __builtin_amdgcn_wmma_f32_16x16x32_bf16(false, al1, false, bh,
                                                       (short)0, acc1, false, false);
    }

    // ---- store C/D: VGPR r -> M = mbase + r, N = lane&15 ----
    float* __restrict__ ob = xv + (size_t)b * 256u * 8192u + p0 + (lane & 15);
    const int mb = m0 + (lane >> 4) * 8;
#pragma unroll
    for (int r = 0; r < 8; ++r) {
        ob[(size_t)(mb + r) * 8192u]      = acc0[r];
        ob[(size_t)(mb + 16 + r) * 8192u] = acc1[r];
    }
}

// ---------------------------------------------------------------------------
// K2: per-channel mean/var over (B,h,w) -> fused affine scale/shift.
// One block per channel; LDS tree reduction.
// ---------------------------------------------------------------------------
__global__ __launch_bounds__(256) void k2_stats(
    const float* __restrict__ xv, const float* __restrict__ gamma,
    const float* __restrict__ beta, float* __restrict__ scale,
    float* __restrict__ shift)
{
    __shared__ float ss[256];
    __shared__ float s2s[256];
    const int c = blockIdx.x;
    float s = 0.f, s2 = 0.f;
    for (int i = threadIdx.x; i < 32768; i += 256) {
        const int b = i >> 13, p = i & 8191;
        float v = xv[(size_t)b * 2097152u + (size_t)c * 8192u + p];
        s += v; s2 += v * v;
    }
    ss[threadIdx.x] = s; s2s[threadIdx.x] = s2;
    __syncthreads();
    for (int st = 128; st > 0; st >>= 1) {
        if (threadIdx.x < st) {
            ss[threadIdx.x]  += ss[threadIdx.x + st];
            s2s[threadIdx.x] += s2s[threadIdx.x + st];
        }
        __syncthreads();
    }
    if (threadIdx.x == 0) {
        const float inv_n = 1.f / 32768.f;
        float mean = ss[0] * inv_n;
        float var  = s2s[0] * inv_n - mean * mean;
        float rs   = rsqrtf(var + 1e-5f);
        float g    = gamma[c] * rs;
        scale[c] = g;
        shift[c] = beta[c] - mean * g;
    }
}

// ---------------------------------------------------------------------------
// K3: 3x3 local-attention softmax weights from guide (q=k), border-masked.
// One thread per output pixel.
// ---------------------------------------------------------------------------
__global__ __launch_bounds__(256) void k3_scores(
    const float* __restrict__ guide, float* __restrict__ attn)
{
    const int pid = blockIdx.x * 256 + threadIdx.x;      // 0..131071
    const int b  = pid >> 15;
    const int hw = pid & 32767;
    const int h  = hw >> 8;
    const int wc = hw & 255;

    const float* __restrict__ gb = guide + (size_t)b * 2097152u; // 64*32768

    bool  valid[9];
    float s[9];
#pragma unroll
    for (int t = 0; t < 9; ++t) {
        const int hh = h + t / 3 - 1, ww = wc + t % 3 - 1;
        valid[t] = (hh >= 0) & (hh < 128) & (ww >= 0) & (ww < 256);
        s[t] = 0.f;
    }
    for (int c = 0; c < 64; ++c) {
        const float* __restrict__ gc = gb + (size_t)c * 32768u;
        const float q = gc[hw];
#pragma unroll
        for (int t = 0; t < 9; ++t) {
            if (valid[t])
                s[t] += q * gc[(h + t / 3 - 1) * 256 + (wc + t % 3 - 1)];
        }
    }
    float mx = -1e30f;
#pragma unroll
    for (int t = 0; t < 9; ++t) if (valid[t] && s[t] > mx) mx = s[t];
    float e[9], sum = 0.f;
#pragma unroll
    for (int t = 0; t < 9; ++t) {
        e[t] = valid[t] ? __expf(s[t] - mx) : 0.f;
        sum += e[t];
    }
    const float inv = 1.f / sum;
    float* __restrict__ ab = attn + (size_t)b * 294912u + hw;    // 9*32768
#pragma unroll
    for (int t = 0; t < 9; ++t) ab[(size_t)t * 32768u] = e[t] * inv;
}

// ---------------------------------------------------------------------------
// K4: out[b,c,h,w] = sum_t attn_t * (scale[c]*bilerp(xv) + shift[c])
// Upsample (2x bilinear, half-pixel centers) fused -> no 128MB intermediate.
// b,c uniform per block (scalar loads). Output is write-once -> NT stores
// so the 128MB result does not evict xv/guide/attn from the 192MB L2.
// ---------------------------------------------------------------------------
__global__ __launch_bounds__(256) void k4_out(
    const float* __restrict__ xv, const float* __restrict__ attn,
    const float* __restrict__ scale, const float* __restrict__ shift,
    float* __restrict__ out)
{
    const size_t o = (size_t)blockIdx.x * 256u + threadIdx.x;
    const int wc = (int)(o & 255);
    const int h  = (int)((o >> 8) & 127);
    const int c  = (int)((o >> 15) & 255);
    const int b  = (int)(o >> 23);

    const float* __restrict__ ab = attn + (size_t)b * 294912u + h * 256 + wc;
    const float* __restrict__ vb = xv + ((size_t)b * 256u + c) * 8192u;
    const float sc = scale[c], sh = shift[c];

    float accv = 0.f, acca = 0.f;
#pragma unroll
    for (int t = 0; t < 9; ++t) {
        const int hh = h + t / 3 - 1, ww = wc + t % 3 - 1;
        if (hh < 0 || hh >= 128 || ww < 0 || ww >= 256) continue;
        const float a = ab[(size_t)t * 32768u];
        // half-pixel-center 2x bilinear source coords in the 64x128 grid
        const float fy = 0.5f * hh - 0.25f;
        const float fx = 0.5f * ww - 0.25f;
        const float y0f = floorf(fy), x0f = floorf(fx);
        const float wy = fy - y0f,  wx = fx - x0f;
        int y0 = (int)y0f, x0 = (int)x0f;
        int y1 = y0 + 1,   x1 = x0 + 1;
        y0 = max(0, min(63, y0));  y1 = max(0, min(63, y1));
        x0 = max(0, min(127, x0)); x1 = max(0, min(127, x1));
        const float v00 = vb[y0 * 128 + x0];
        const float v01 = vb[y0 * 128 + x1];
        const float v10 = vb[y1 * 128 + x0];
        const float v11 = vb[y1 * 128 + x1];
        const float v = (1.f - wy) * ((1.f - wx) * v00 + wx * v01)
                      +        wy  * ((1.f - wx) * v10 + wx * v11);
        accv += a * v;
        acca += a;
    }
    __builtin_nontemporal_store(sc * accv + sh * acca, &out[o]);
}

// ---------------------------------------------------------------------------
extern "C" void kernel_launch(void* const* d_in, const int* in_sizes, int n_in,
                              void* d_out, int out_size, void* d_ws, size_t ws_size,
                              hipStream_t stream)
{
    const float* x     = (const float*)d_in[0];
    const float* guide = (const float*)d_in[1];
    const float* w     = (const float*)d_in[2];
    const float* gamma = (const float*)d_in[3];
    const float* beta  = (const float*)d_in[4];
    float* out = (float*)d_out;

    float* ws    = (float*)d_ws;
    float* xv    = ws;                      // 8,388,608 floats
    float* scale = ws + 8388608u;           // 256
    float* shift = ws + 8388864u;           // 256
    float* attn  = ws + 8389120u;           // 1,179,648

    // K1: WMMA GEMM (2048 n-tiles; 8 waves/block, 2 m-tiles/wave, async DMA)
    k1_gemm_wmma<<<dim3(2048), 256, 0, stream>>>(x, w, xv);
    // K2: per-channel stats -> scale/shift
    k2_stats<<<256, 256, 0, stream>>>(xv, gamma, beta, scale, shift);
    // K3: attention weights (131072 pixels)
    k3_scores<<<512, 256, 0, stream>>>(guide, attn);
    // K4: fused upsample + weighted sum (33,554,432 outputs)
    k4_out<<<131072, 256, 0, stream>>>(xv, attn, scale, shift, out);
}
